// lstm_30734785970219
// MI455X (gfx1250) — matrix operand order repaired and checked by reading, not compiled, not run on hardware
//
#include <hip/hip_runtime.h>
#include <hip/hip_bf16.h>

// ---------------------------------------------------------------------------
// LSTM cell (non-recurrent, per-position):
//   z = [x|h] @ [Wf|Wi|Wg|Wo] + b ; gates ; c_out = f*c + g*i ; h_out = tanh(c)*o
// M = 4096, K = 2048, N = 4*1024.  bf16 WMMA GEMM, f32 accumulate.
// Staging: Tensor Data Mover (tensor_load_to_lds) double-buffered into padded
// LDS tiles; fused register-resident gate epilogue (all 4 gates per wave).
// ---------------------------------------------------------------------------

typedef __attribute__((ext_vector_type(16))) __bf16 v16bf;
typedef __attribute__((ext_vector_type(8)))  __bf16 v8bf;
typedef __attribute__((ext_vector_type(4)))  __bf16 v4bf;
typedef __attribute__((ext_vector_type(8)))  float  v8f;
typedef __attribute__((ext_vector_type(4)))  float  f32x4;
typedef __attribute__((ext_vector_type(4)))  unsigned int u32x4;
typedef __attribute__((ext_vector_type(8)))  int    i32x8;
typedef __attribute__((ext_vector_type(4)))  int    i32x4;

#define M_TOT 4096      // B*T
#define K_TOT 2048      // D_IN + H
#define H_TOT 1024
#define BM    128       // M rows per workgroup
#define BH    64        // H cols per workgroup (x4 gates)
#define KT    32        // K step (one bf16 WMMA depth)
#define LDK   40        // padded LDS K-stride in bf16 elems (80B, conflict-free)

union Frag16 { v16bf v; v8bf h[2]; };

__device__ __forceinline__ float fast_sigmoid(float x) {
    return 1.f / (1.f + __expf(-x));
}
// branch-free tanh: 1 - 2/(e^{2x}+1); correct limits at +/-inf
__device__ __forceinline__ float fast_tanh(float x) {
    return 1.f - 2.f / (__expf(2.f * x) + 1.f);
}

// --- pack [x | h_in] -> bf16 A[M][K] ---------------------------------------
__global__ void pack_a(const float* __restrict__ x, const float* __restrict__ hin,
                       __bf16* __restrict__ A) {
    int t    = blockIdx.x * blockDim.x + threadIdx.x;   // 4 elems / thread
    int base = t * 4;
    int m    = base >> 11;          // /K_TOT
    int k    = base & (K_TOT - 1);
    const float* src = (k < H_TOT) ? (x + (size_t)m * H_TOT + k)
                                   : (hin + (size_t)m * H_TOT + (k - H_TOT));
    f32x4 v = *(const f32x4*)src;
    v4bf  o;
    o[0] = (__bf16)v[0]; o[1] = (__bf16)v[1];
    o[2] = (__bf16)v[2]; o[3] = (__bf16)v[3];
    *(v4bf*)&A[base] = o;
}

// --- transpose+convert W_g[K][H] -> bf16 Bt[g*1024 + n][K] -----------------
__global__ void pack_bt(const float* __restrict__ Wf, const float* __restrict__ Wi,
                        const float* __restrict__ Wg, const float* __restrict__ Wo,
                        __bf16* __restrict__ Bt) {
    __shared__ float tile[32][33];
    const int g = blockIdx.z;
    const float* W = (g == 0) ? Wf : (g == 1) ? Wi : (g == 2) ? Wg : Wo;
    const int k0 = blockIdx.x * 32;
    const int c0 = blockIdx.y * 32;
    const int tx = threadIdx.x, ty = threadIdx.y;      // 32 x 8
    #pragma unroll
    for (int r = 0; r < 32; r += 8)
        tile[ty + r][tx] = W[(size_t)(k0 + ty + r) * H_TOT + c0 + tx];
    __syncthreads();
    #pragma unroll
    for (int r = 0; r < 32; r += 8) {
        int n = g * H_TOT + c0 + ty + r;
        Bt[(size_t)n * K_TOT + k0 + tx] = (__bf16)tile[tx][ty + r];
    }
}

// --- TDM: DMA a (padded) tile into LDS -------------------------------------
// D# per CDNA5 ISA §8.3-8.6.  data_size=2B; pad_enable with pad_interval=16
// DWORDs (one 64B tile row) and pad_amount=4 DWORDs (16B) -> LDS stride 80B.
__device__ __forceinline__ void tdm_load(
    unsigned lds_addr, unsigned long long gaddr,
    unsigned tensor_d0, unsigned tensor_d1,
    unsigned tile_d0, unsigned tile_d1, unsigned tile_d2,
    unsigned long long d0_stride, unsigned long long d1_stride,
    unsigned tensor_d2)
{
    u32x4 g0;
    g0[0] = 1u;                                        // count=1, user mode
    g0[1] = lds_addr;                                  // LDS byte address
    g0[2] = (unsigned)gaddr;                           // global addr [31:0]
    g0[3] = (unsigned)(gaddr >> 32) | (2u << 30);      // [56:32] | type=2

    i32x8 g1;
    g1[0] = (int)((1u << 16)        // data_size = 2 bytes
                | (1u << 20)        // pad_enable
                | (3u << 22)        // pad_interval: 16 DWORDs
                | (3u << 25));      // pad_amount:   4 DWORDs
    g1[1] = (int)((tensor_d0 & 0xFFFFu) << 16);
    g1[2] = (int)(((tensor_d0 >> 16) & 0xFFFFu) | ((tensor_d1 & 0xFFFFu) << 16));
    g1[3] = (int)(((tensor_d1 >> 16) & 0xFFFFu) | (tile_d0 << 16));
    g1[4] = (int)((tile_d1 & 0xFFFFu) | (tile_d2 << 16));
    g1[5] = (int)(unsigned)d0_stride;
    g1[6] = (int)((unsigned)((d0_stride >> 32) & 0xFFFFu)
                | (((unsigned)d1_stride & 0xFFFFu) << 16));
    g1[7] = (int)(unsigned)(d1_stride >> 16);

    i32x4 g2;
    g2[0] = (int)tensor_d2;   // tensor_dim2 (OOB bound for dim2)
    g2[1] = 0;                // tensor_dim3
    g2[2] = 0;                // tensor_dim2_stride lo
    g2[3] = 0;                // tensor_dim2_stride hi | tile_dim3
    i32x4 g3;
    g3[0] = 0; g3[1] = 0; g3[2] = 0; g3[3] = 0;

    i32x8 g4;                 // reserved/unused group (zero-filled)
    g4[0] = 0; g4[1] = 0; g4[2] = 0; g4[3] = 0;
    g4[4] = 0; g4[5] = 0; g4[6] = 0; g4[7] = 0;

    __builtin_amdgcn_tensor_load_to_lds(g0, g1, g2, g3, g4, 0);
}

// --- main fused GEMM + gate kernel -----------------------------------------
__global__ __launch_bounds__(256)
void lstm_wmma(const __bf16* __restrict__ A, const __bf16* __restrict__ Bt,
               const float* __restrict__ cin,
               const float* __restrict__ bF, const float* __restrict__ bI,
               const float* __restrict__ bG, const float* __restrict__ bO,
               float* __restrict__ out) {
    __shared__ __align__(16) __bf16 As[2][BM][LDK];        // 20480 B
    __shared__ __align__(16) __bf16 Bs[2][4][BH][LDK];     // 40960 B

    const int mBase   = blockIdx.x * BM;
    const int hBase   = blockIdx.y * BH;
    const int t       = threadIdx.x;
    const int lane    = t & 31;
    const int laneRow = lane & 15;
    const int hi      = lane >> 4;          // 0 = lanes 0-15, 1 = lanes 16-31

    // wave id as a *uniform* scalar -> scalar branches around TDM issue
    const int waveId  = __builtin_amdgcn_readfirstlane(t) >> 5;

    // wave -> 2 M-tiles x 2 H-tiles x 4 gates (16 WMMA tiles, 128 acc VGPRs)
    const int mSel = (waveId & 3) * 32;     // wave's 32 M-rows within block
    const int hSel = (waveId >> 2) * 32;    // wave's 32 H-cols within block

    v8f acc[4][2][2];                        // [gate][tn][tm]
    #pragma unroll
    for (int g = 0; g < 4; ++g)
        #pragma unroll
        for (int tn = 0; tn < 2; ++tn)
            #pragma unroll
            for (int tm = 0; tm < 2; ++tm)
                #pragma unroll
                for (int e = 0; e < 8; ++e) acc[g][tn][tm][e] = 0.f;

    // LDS byte addresses (flat shared addr truncates to LDS offset)
    const unsigned ldsA0 = (unsigned)(unsigned long long)(void*)&As[0][0][0];
    const unsigned ldsA1 = (unsigned)(unsigned long long)(void*)&As[1][0][0];
    const unsigned ldsB0 = (unsigned)(unsigned long long)(void*)&Bs[0][0][0][0];
    const unsigned ldsB1 = (unsigned)(unsigned long long)(void*)&Bs[1][0][0][0];

    const unsigned long long aBase =
        (unsigned long long)(const void*)A + (unsigned long long)mBase * K_TOT * 2;
    const unsigned long long bBase =
        (unsigned long long)(const void*)Bt + (unsigned long long)hBase * K_TOT * 2;

    auto issueA = [&](int buf, int kt) {
        tdm_load(buf ? ldsA1 : ldsA0, aBase + (unsigned)kt * 2,
                 /*tensor_d0=*/K_TOT, /*tensor_d1=*/M_TOT,
                 /*tile_d0=*/KT, /*tile_d1=*/BM, /*tile_d2=*/0,
                 /*d0_stride=*/K_TOT, /*d1_stride=*/0, /*tensor_d2=*/0);
    };
    auto issueB = [&](int buf, int kt) {
        // 3D: dim0=K chunk, dim1=BH rows, dim2=4 gates (gate stride 1024 rows)
        tdm_load(buf ? ldsB1 : ldsB0, bBase + (unsigned)kt * 2,
                 /*tensor_d0=*/K_TOT, /*tensor_d1=*/M_TOT,
                 /*tile_d0=*/KT, /*tile_d1=*/BH, /*tile_d2=*/4,
                 /*d0_stride=*/K_TOT,
                 /*d1_stride=*/(unsigned long long)H_TOT * K_TOT,
                 /*tensor_d2=*/4);
    };

    if (waveId == 0) issueA(0, 0);
    if (waveId == 1) issueB(0, 0);
    __builtin_amdgcn_s_wait_tensorcnt(0);   // no-op for waves with TENSORcnt==0
    __syncthreads();

    for (int kt = 0; kt < K_TOT; kt += KT) {
        const int buf = (kt >> 5) & 1;
        if (kt + KT < K_TOT) {
            if (waveId == 0) issueA(buf ^ 1, kt + KT);
            if (waveId == 1) issueB(buf ^ 1, kt + KT);
        }

        // A fragments (16x32 bf16): lane<16 holds K {0..7, 16..23}
        Frag16 af[2];
        #pragma unroll
        for (int tm = 0; tm < 2; ++tm) {
            const __bf16* p = &As[buf][mSel + tm * 16 + laneRow][0];
            const int k0 = hi * 8;
            af[tm].h[0] = *(const v8bf*)(p + k0);
            af[tm].h[1] = *(const v8bf*)(p + k0 + 16);
        }
        #pragma unroll
        for (int g = 0; g < 4; ++g) {
            #pragma unroll
            for (int tn = 0; tn < 2; ++tn) {
                // B fragment (32x16): lane<16 holds K {0..7, 16..23}
                Frag16 bf;
                const __bf16* p = &Bs[buf][g][hSel + tn * 16 + laneRow][0];
                const int k0 = hi * 8;
                bf.h[0] = *(const v8bf*)(p + k0);
                bf.h[1] = *(const v8bf*)(p + k0 + 16);
                #pragma unroll
                for (int tm = 0; tm < 2; ++tm)
                    acc[g][tn][tm] = __builtin_amdgcn_wmma_f32_16x16x32_bf16(
                        false, af[tm].v, false, bf.v, (short)0,
                        acc[g][tn][tm], false, false);
            }
        }

        __builtin_amdgcn_s_wait_tensorcnt(0);
        __syncthreads();
    }

    // ---- register-resident gate epilogue ----
    // C layout: VGPR v, lanes 0-15 -> M=v, N=lane; lanes 16-31 -> M=v+8.
    #pragma unroll
    for (int tn = 0; tn < 2; ++tn) {
        const int h = hBase + hSel + tn * 16 + laneRow;
        const float zbF = bF[h], zbI = bI[h], zbG = bG[h], zbO = bO[h];
        #pragma unroll
        for (int tm = 0; tm < 2; ++tm) {
            #pragma unroll
            for (int v = 0; v < 8; ++v) {
                const int m = mBase + mSel + tm * 16 + v + hi * 8;
                const float zf = acc[0][tn][tm][v] + zbF;
                const float zi = acc[1][tn][tm][v] + zbI;
                const float zg = acc[2][tn][tm][v] + zbG;
                const float zo = acc[3][tn][tm][v] + zbO;
                const float fg = fast_sigmoid(zf);
                const float ig = fast_sigmoid(zi);
                const float gg = fast_tanh(zg);
                const float og = fast_sigmoid(zo);
                const float c  = cin[(size_t)m * H_TOT + h];
                const float cOut = fg * c + gg * ig;
                const float hOut = fast_tanh(cOut) * og;
                out[(size_t)m * H_TOT + h] = hOut;                              // h_out
                out[(size_t)M_TOT * H_TOT + (size_t)m * H_TOT + h] = cOut;      // c_out
            }
        }
    }
}

extern "C" void kernel_launch(void* const* d_in, const int* in_sizes, int n_in,
                              void* d_out, int out_size, void* d_ws, size_t ws_size,
                              hipStream_t stream) {
    (void)in_sizes; (void)n_in; (void)out_size; (void)ws_size;
    const float* x   = (const float*)d_in[0];
    const float* hin = (const float*)d_in[1];
    const float* cin = (const float*)d_in[2];
    const float* Wf  = (const float*)d_in[3];
    const float* bf  = (const float*)d_in[4];
    const float* Wi  = (const float*)d_in[5];
    const float* bi  = (const float*)d_in[6];
    const float* Wg  = (const float*)d_in[7];
    const float* bg  = (const float*)d_in[8];
    const float* Wo  = (const float*)d_in[9];
    const float* bo  = (const float*)d_in[10];

    __bf16* Abuf  = (__bf16*)d_ws;                                        // 16 MB
    __bf16* Btbuf = (__bf16*)((char*)d_ws + (size_t)M_TOT * K_TOT * 2);   // 16 MB
    float*  out   = (float*)d_out;

    pack_a<<<(M_TOT * (size_t)K_TOT / 4) / 256, 256, 0, stream>>>(x, hin, Abuf);
    pack_bt<<<dim3(K_TOT / 32, H_TOT / 32, 4), dim3(32, 8), 0, stream>>>(Wf, Wi, Wg, Wo, Btbuf);
    lstm_wmma<<<dim3(M_TOT / BM, H_TOT / BH), 256, 0, stream>>>(
        Abuf, Btbuf, cin, bf, bi, bg, bo, out);
}
